// GlmImageAttention_76081050681760
// MI455X (gfx1250) — matrix operand order repaired and checked by reading, not compiled, not run on hardware
//
#include <hip/hip_runtime.h>
#include <hip/hip_bf16.h>
#include <math.h>

// ---------------------------------------------------------------------------
// CDNA5 (gfx1250) wave32 WMMA implementation of GLM image attention.
// All matmuls via v_wmma_f32_16x16x32_f16 (f16 in, f32 accumulate).
// Hot loops stage tiles with GLOBAL_LOAD_ASYNC_TO_LDS_B128 (ASYNCcnt-tracked,
// no VGPR staging -> no scratch spills) into double-buffered LDS; one
// s_wait_asynccnt + barrier per K-step. Falls back to register staging if
// the async builtins are unavailable.
// ---------------------------------------------------------------------------

#if __has_builtin(__builtin_amdgcn_global_load_async_to_lds_b128)
#define USE_ASYNC_LDS 1
#else
#define USE_ASYNC_LDS 0
#endif

typedef _Float16 v16h __attribute__((ext_vector_type(16)));
typedef float    v8f  __attribute__((ext_vector_type(8)));
typedef int      vi4  __attribute__((ext_vector_type(4)));
typedef __attribute__((address_space(1))) vi4 gvi4;  // global int4
typedef __attribute__((address_space(3))) vi4 lvi4;  // LDS int4

union Frag16 { v16h v; unsigned u[8]; };

namespace cfg {
constexpr int B    = 2;
constexpr int SEQ  = 1280;      // 256 text + 1024 image
constexpr int TXT  = 256;
constexpr int IMG  = SEQ - TXT; // 1024
constexpr int H    = 24;
constexpr int DH   = 128;
constexpr int DM   = 3072;
constexpr int N3   = 3 * DM;    // 9216
constexpr int ROWS = B * SEQ;   // 2560
}

__device__ __forceinline__ void async_copy_b128(const unsigned* g, unsigned* l)
{
#if USE_ASYNC_LDS
    __builtin_amdgcn_global_load_async_to_lds_b128((gvi4*)g, (lvi4*)l, 0, 0);
#else
    (void)g; (void)l;
#endif
}

__device__ __forceinline__ void wait_async0()
{
#if __has_builtin(__builtin_amdgcn_s_wait_asynccnt)
    __builtin_amdgcn_s_wait_asynccnt(0);
#else
    asm volatile("s_wait_asynccnt 0" ::: "memory");
#endif
}

__device__ __forceinline__ unsigned packf16(float lo, float hi)
{
    union { _Float16 h[2]; unsigned u; } t;
    t.h[0] = (_Float16)lo; t.h[1] = (_Float16)hi;
    return t.u;
}

// ---------------------------------------------------------------------------
// Pack W[K][N] f32 -> WtP[N][K/2] dwords; dword p at column n holds f16
// halves (K=2p, K=2p+1) -- exactly one B-matrix fragment dword.
// ---------------------------------------------------------------------------
__global__ __launch_bounds__(256) void pack_w_kernel(
    const float* __restrict__ W, unsigned* __restrict__ WtP, int K, int N)
{
    constexpr int KP = 16;  // kpairs per tile
    constexpr int NW = 64;  // n per tile
    __shared__ unsigned T[NW * KP];  // [n][p]
    const int nkp = K / 2 / KP;
    const int kp0 = (blockIdx.x % nkp) * KP;
    const int n0  = (blockIdx.x / nkp) * NW;
    const int tid = threadIdx.x;
    {
        int p = tid >> 4, n4 = (tid & 15) * 4;
        const float* r0 = W + (size_t)(2 * (kp0 + p)) * N + n0 + n4;
        float a[4], b4[4];
        *(float4*)a  = *(const float4*)r0;
        *(float4*)b4 = *(const float4*)(r0 + N);
#pragma unroll
        for (int j = 0; j < 4; ++j) T[(n4 + j) * KP + p] = packf16(a[j], b4[j]);
    }
    __syncthreads();
    {
        int n = tid >> 2, q = (tid & 3) * 4;
        *(uint4*)(WtP + (size_t)(n0 + n) * (K / 2) + kp0 + q) =
            *(const uint4*)&T[n * KP + q];
    }
}

// ---------------------------------------------------------------------------
// Pack X = concat(text, image) rows, f32 -> f16 [ROWS][DM].
// ---------------------------------------------------------------------------
__global__ __launch_bounds__(256) void pack_x_kernel(
    const float* __restrict__ enc, const float* __restrict__ hid,
    _Float16* __restrict__ Xh)
{
    size_t idx = (size_t)blockIdx.x * 256 + threadIdx.x;  // units of 4 cols
    constexpr int PR = cfg::DM / 4;
    int g  = (int)(idx / PR);
    int c4 = (int)(idx % PR) * 4;
    int b = g / cfg::SEQ, s = g % cfg::SEQ;
    const float* src = (s < cfg::TXT)
        ? enc + ((size_t)b * cfg::TXT + s) * cfg::DM + c4
        : hid + ((size_t)b * cfg::IMG + (s - cfg::TXT)) * cfg::DM + c4;
    float a[4];
    *(float4*)a = *(const float4*)src;
    union { _Float16 h[4]; uint2 u; } o;
#pragma unroll
    for (int i = 0; i < 4; ++i) o.h[i] = (_Float16)a[i];
    *(uint2*)(Xh + (size_t)g * cfg::DM + c4) = o.u;
}

// ---------------------------------------------------------------------------
// WMMA GEMM: out[M,N] = Ah[M,K](f16) * W(WtP pair-packed) + bias.
// 128x128 tile / 256-thread block; 8 waves in 4(M) x 2(N); each wave owns
// 32x64 = 2x4 WMMA tiles. K staged in chunks of 32 via async-to-LDS into a
// ping-pong buffer; s_wait_asynccnt + barrier per step.
//   outmode==0: f16 store row-major; outmode==1: f32 store, output row remap
// ---------------------------------------------------------------------------
__global__ __launch_bounds__(256) void wmma_gemm_f16(
    const _Float16* __restrict__ Ah,
    const unsigned* __restrict__ WtP,
    const float* __restrict__ bias,
    void* __restrict__ outp,
    int K, int N, int outmode)
{
    constexpr int BM = 128, BN = 128, BK = 32;
    constexpr int LDA = 20;  // dwords per As row (16 + 4 pad)
    constexpr int LDW = 20;  // dwords per Ws row (16 + 4 pad)
    __shared__ unsigned As[2][BM * LDA];  // [m][kpair], double buffered
    __shared__ unsigned Ws[2][BN * LDW];  // [n][kpair], double buffered

    const int tid  = threadIdx.x;
    const int lane = tid & 31;
    const int wave = tid >> 5;
    const int hi   = (lane >> 4) & 1;
    const int ln   = lane & 15;
    const int m0   = blockIdx.y * BM;
    const int n0   = blockIdx.x * BN;
    const int mw   = (wave >> 1) * 32;  // 0/32/64/96
    const int nw   = (wave & 1) * 64;   // 0/64

    const unsigned* AhU = (const unsigned*)Ah;
    const int K2 = K / 2;
    // per-thread staging coordinates (2 b128 units each for A and W)
    const int sr = tid >> 2;            // 0..63 (+64 for unit 1)
    const int sq = (tid & 3) * 4;

    v8f acc[2][4];
#pragma unroll
    for (int i = 0; i < 2; ++i)
#pragma unroll
        for (int j = 0; j < 4; ++j)
#pragma unroll
            for (int e = 0; e < 8; ++e) acc[i][j][e] = 0.0f;

#if USE_ASYNC_LDS
    auto stage_tile = [&](int buf, int kp) {
#pragma unroll
        for (int i = 0; i < 2; ++i)
            async_copy_b128(AhU + (size_t)(m0 + sr + i * 64) * K2 + kp + sq,
                            &As[buf][(sr + i * 64) * LDA + sq]);
#pragma unroll
        for (int i = 0; i < 2; ++i)
            async_copy_b128(WtP + (size_t)(n0 + sr + i * 64) * K2 + kp + sq,
                            &Ws[buf][(sr + i * 64) * LDW + sq]);
    };
#else
    uint4 ra[2], rw[2];
    auto load_tile = [&](int kp) {
#pragma unroll
        for (int i = 0; i < 2; ++i)
            ra[i] = *(const uint4*)(AhU + (size_t)(m0 + sr + i * 64) * K2 + kp + sq);
#pragma unroll
        for (int i = 0; i < 2; ++i)
            rw[i] = *(const uint4*)(WtP + (size_t)(n0 + sr + i * 64) * K2 + kp + sq);
    };
    auto store_tile = [&](int buf) {
#pragma unroll
        for (int i = 0; i < 2; ++i)
            *(uint4*)&As[buf][(sr + i * 64) * LDA + sq] = ra[i];
#pragma unroll
        for (int i = 0; i < 2; ++i)
            *(uint4*)&Ws[buf][(sr + i * 64) * LDW + sq] = rw[i];
    };
#endif

    auto do_wmma = [&](int buf) {
        const unsigned* cA = &As[buf][0];
        const unsigned* cW = &Ws[buf][0];
        Frag16 a[2], b[4];
#pragma unroll
        for (int mt = 0; mt < 2; ++mt) {
            int rb = (mw + mt * 16 + ln) * LDA;
#pragma unroll
            for (int vg = 0; vg < 8; ++vg) {
                int kd = vg + hi * 8;
                a[mt].u[vg] = cA[rb + kd];
            }
        }
#pragma unroll
        for (int nt = 0; nt < 4; ++nt) {
            int rb = (nw + nt * 16 + ln) * LDW;
#pragma unroll
            for (int vg = 0; vg < 8; ++vg)
                b[nt].u[vg] = cW[rb + vg + hi * 8];
        }
#pragma unroll
        for (int mt = 0; mt < 2; ++mt)
#pragma unroll
            for (int nt = 0; nt < 4; ++nt)
                acc[mt][nt] = __builtin_amdgcn_wmma_f32_16x16x32_f16(
                    false, a[mt].v, false, b[nt].v, (short)0, acc[mt][nt],
                    false, false);
    };

    const int nk = K / BK;
#if USE_ASYNC_LDS
    stage_tile(0, 0);
    for (int ks = 0; ks < nk; ++ks) {
        wait_async0();     // my async copies into buf[ks&1] (and older) done
        __syncthreads();   // everyone's copies visible; prev reads finished
        if (ks + 1 < nk) {
            stage_tile((ks + 1) & 1, (ks + 1) * (BK / 2));
            int kp2 = ((ks + 2 < nk) ? (ks + 2) : (nk - 1)) * (BK / 2);
            __builtin_prefetch(WtP + (size_t)(n0 + (tid >> 1)) * K2 + kp2, 0, 1);
        }
        do_wmma(ks & 1);
    }
#else
    load_tile(0);
    store_tile(0);
    for (int ks = 0; ks < nk - 1; ++ks) {
        __syncthreads();
        load_tile((ks + 1) * (BK / 2));
        do_wmma(ks & 1);
        store_tile((ks + 1) & 1);
    }
    __syncthreads();
    do_wmma((nk - 1) & 1);
#endif

    // ---- epilogue: C layout row = r + 8*hi, col = lane%16 ----
#pragma unroll
    for (int mt = 0; mt < 2; ++mt)
#pragma unroll
        for (int nt = 0; nt < 4; ++nt)
#pragma unroll
            for (int r = 0; r < 8; ++r) {
                int grow = m0 + mw + mt * 16 + r + hi * 8;
                int col  = n0 + nw + nt * 16 + ln;
                float vv = acc[mt][nt][r] + bias[col];
                if (outmode == 0) {
                    ((_Float16*)outp)[(size_t)grow * N + col] = (_Float16)vv;
                } else {
                    int b = grow / cfg::SEQ, s = grow % cfg::SEQ;
                    size_t drow = (s >= cfg::TXT)
                        ? ((size_t)b * cfg::IMG + (s - cfg::TXT))
                        : ((size_t)cfg::B * cfg::IMG + (size_t)b * cfg::TXT + s);
                    ((float*)outp)[drow * N + col] = vv;
                }
            }
}

// ---------------------------------------------------------------------------
// LayerNorm (no affine) + partial RoPE + repack. One wave per (b,s,h).
// ---------------------------------------------------------------------------
__device__ __forceinline__ void ln128(float x[4])
{
    float ssum = x[0] + x[1] + x[2] + x[3];
#pragma unroll
    for (int m = 1; m < 32; m <<= 1) ssum += __shfl_xor(ssum, m, 32);
    float mu = ssum * (1.0f / 128.0f);
    float e[4]; float s2 = 0.f;
#pragma unroll
    for (int i = 0; i < 4; ++i) { e[i] = x[i] - mu; s2 += e[i] * e[i]; }
#pragma unroll
    for (int m = 1; m < 32; m <<= 1) s2 += __shfl_xor(s2, m, 32);
    float r = rsqrtf(s2 * (1.0f / 128.0f) + 1e-5f);
#pragma unroll
    for (int i = 0; i < 4; ++i) x[i] = e[i] * r;
}

__global__ __launch_bounds__(256) void ln_rope_pack_kernel(
    const _Float16* __restrict__ qkv,
    const float* __restrict__ cosb, const float* __restrict__ sinb,
    _Float16* __restrict__ Qo, _Float16* __restrict__ Ko,
    _Float16* __restrict__ Vt)
{
    int wid  = blockIdx.x * 8 + (threadIdx.x >> 5);
    int lane = threadIdx.x & 31;
    int b = wid / (cfg::SEQ * cfg::H);
    int rem = wid % (cfg::SEQ * cfg::H);
    int s = rem / cfg::H;
    int h = rem % cfg::H;
    const _Float16* rowp = qkv + (size_t)(b * cfg::SEQ + s) * cfg::N3;
    int d0 = lane * 4;

    union U4 { uint2 u; _Float16 h4[4]; } tq, tk, tv;
    tq.u = *(const uint2*)(rowp + h * cfg::DH + d0);
    tk.u = *(const uint2*)(rowp + cfg::DM + h * cfg::DH + d0);
    tv.u = *(const uint2*)(rowp + 2 * cfg::DM + h * cfg::DH + d0);
    float q[4], k[4], v[4];
#pragma unroll
    for (int i = 0; i < 4; ++i) {
        q[i] = (float)tq.h4[i]; k[i] = (float)tk.h4[i]; v[i] = (float)tv.h4[i];
    }

    ln128(q);
    ln128(k);

    if (s >= cfg::TXT) {  // RoPE on image tokens only
        int idx = s - cfg::TXT;
        float c4[4], s4[4];
        *(float4*)c4 = *(const float4*)(cosb + (size_t)idx * cfg::DH + d0);
        *(float4*)s4 = *(const float4*)(sinb + (size_t)idx * cfg::DH + d0);
        float sgn = (d0 < 64) ? -1.f : 1.f;  // rot = [-imag, real]
#pragma unroll
        for (int i = 0; i < 4; ++i) {
            float qp = __shfl_xor(q[i], 16, 32);  // partner at d +/- 64
            float kp = __shfl_xor(k[i], 16, 32);
            q[i] = q[i] * c4[i] + sgn * qp * s4[i];
            k[i] = k[i] * c4[i] + sgn * kp * s4[i];
        }
    }

    union U4 pq, pk;
#pragma unroll
    for (int i = 0; i < 4; ++i) { pq.h4[i] = (_Float16)q[i]; pk.h4[i] = (_Float16)k[i]; }
    size_t base = ((size_t)(b * cfg::H + h) * cfg::SEQ + s) * cfg::DH + d0;
    *(uint2*)(Qo + base) = pq.u;
    *(uint2*)(Ko + base) = pk.u;
#pragma unroll
    for (int i = 0; i < 4; ++i)
        Vt[((size_t)(b * cfg::H + h) * cfg::DH + d0 + i) * cfg::SEQ + s] =
            (_Float16)v[i];
}

// ---------------------------------------------------------------------------
// Flash attention. 8 waves/block; each wave owns a 16-row Q tile (block = 128
// Q rows); KV in 32-row chunks, async-staged into ping-pong LDS.
// P round-trips a per-wave LDS tile (wave-local DS ordering, no barrier).
// 16 WMMA/step/wave.
// ---------------------------------------------------------------------------
__global__ __launch_bounds__(256) void attn_fa_kernel(
    const _Float16* __restrict__ Q, const _Float16* __restrict__ Kd,
    const _Float16* __restrict__ Vt, _Float16* __restrict__ AO)
{
    constexpr int KT  = 32;
    constexpr int LDK = 68;  // dwords per K row (64 + 4)
    constexpr int LDV = 20;  // dwords per Vt row (16 + 4)
    constexpr int LDP = 40;  // halves per P row (32 + 8)
    __shared__ unsigned Klds[2][KT * LDK];
    __shared__ unsigned Vlds[2][cfg::DH * LDV];
    __shared__ _Float16 Plds[8 * 16 * LDP];

    const int tid  = threadIdx.x;
    const int lane = tid & 31;
    const int wave = tid >> 5;
    const int hi   = (lane >> 4) & 1;
    const int ln   = lane & 15;
    constexpr int NQ = cfg::SEQ / 128;  // 10
    const int bh = blockIdx.x / NQ;
    const int qt = blockIdx.x % NQ;
    const int qbase = qt * 128 + wave * 16;

    Frag16 qf[4];
    {
        const unsigned* Qu =
            (const unsigned*)Q + ((size_t)bh * cfg::SEQ + qbase + ln) * (cfg::DH / 2);
#pragma unroll
        for (int kc = 0; kc < 4; ++kc)
#pragma unroll
            for (int vg = 0; vg < 8; ++vg) {
                int kd = vg + hi * 8;
                qf[kc].u[vg] = Qu[kc * 16 + kd];
            }
    }

    v8f O[8];
#pragma unroll
    for (int n = 0; n < 8; ++n)
#pragma unroll
        for (int e = 0; e < 8; ++e) O[n][e] = 0.0f;
    float rowM[8], rowL[8];
#pragma unroll
    for (int r = 0; r < 8; ++r) { rowM[r] = -1e30f; rowL[r] = 0.0f; }

    const float scale = 0.08838834764831845f;  // 1/sqrt(128)
    const unsigned* Kg0 = (const unsigned*)Kd + (size_t)bh * cfg::SEQ * (cfg::DH / 2);
    const unsigned* Vg0 = (const unsigned*)Vt + (size_t)bh * cfg::DH * (cfg::SEQ / 2);
    // staging coordinates (2 b128 units each for K and V, 256 threads)
    const int kr = tid >> 4, kc4 = (tid & 15) * 4;  // K rows: +16 per unit
    const int vd = tid >> 2, vc4 = (tid & 3) * 4;   // V rows: +64 per unit

#if USE_ASYNC_LDS
    auto stage_kv = [&](int buf, int kv0) {
#pragma unroll
        for (int i = 0; i < 2; ++i)
            async_copy_b128(Kg0 + (size_t)(kv0 + kr + i * 16) * 64 + kc4,
                            &Klds[buf][(kr + i * 16) * LDK + kc4]);
#pragma unroll
        for (int i = 0; i < 2; ++i)
            async_copy_b128(Vg0 + (size_t)(vd + i * 64) * (cfg::SEQ / 2) +
                                (kv0 >> 1) + vc4,
                            &Vlds[buf][(vd + i * 64) * LDV + vc4]);
    };
#else
    uint4 rk[2], rv[2];
    auto load_kv = [&](int kv0) {
#pragma unroll
        for (int i = 0; i < 2; ++i)
            rk[i] = *(const uint4*)(Kg0 + (size_t)(kv0 + kr + i * 16) * 64 + kc4);
#pragma unroll
        for (int i = 0; i < 2; ++i)
            rv[i] = *(const uint4*)(Vg0 + (size_t)(vd + i * 64) * (cfg::SEQ / 2) +
                                    (kv0 >> 1) + vc4);
    };
    auto store_kv = [&](int buf) {
#pragma unroll
        for (int i = 0; i < 2; ++i)
            *(uint4*)&Klds[buf][(kr + i * 16) * LDK + kc4] = rk[i];
#pragma unroll
        for (int i = 0; i < 2; ++i)
            *(uint4*)&Vlds[buf][(vd + i * 64) * LDV + vc4] = rv[i];
    };
#endif

    auto compute_step = [&](int buf) {
        const unsigned* cK = &Klds[buf][0];
        const unsigned* cV = &Vlds[buf][0];

        // ---- scores: two 16x16 tiles, K-dim = 128 (4 WMMA each) ----
        v8f c[2];
#pragma unroll
        for (int nt = 0; nt < 2; ++nt) {
            v8f cc;
#pragma unroll
            for (int e = 0; e < 8; ++e) cc[e] = 0.0f;
            int rb = (nt * 16 + ln) * LDK;
#pragma unroll
            for (int kc = 0; kc < 4; ++kc) {
                Frag16 bf;
#pragma unroll
                for (int vg = 0; vg < 8; ++vg)
                    bf.u[vg] = cK[rb + kc * 16 + vg + hi * 8];
                cc = __builtin_amdgcn_wmma_f32_16x16x32_f16(
                    false, qf[kc].v, false, bf.v, (short)0, cc, false, false);
            }
            c[nt] = cc;
        }
#pragma unroll
        for (int nt = 0; nt < 2; ++nt)
#pragma unroll
            for (int r = 0; r < 8; ++r) c[nt][r] *= scale;

        // ---- online softmax ----
        float al[8];
#pragma unroll
        for (int r = 0; r < 8; ++r) {
            float m = fmaxf(c[0][r], c[1][r]);
#pragma unroll
            for (int msk = 1; msk < 16; msk <<= 1)
                m = fmaxf(m, __shfl_xor(m, msk, 32));
            float nm = fmaxf(rowM[r], m);
            al[r] = __expf(rowM[r] - nm);
            rowM[r] = nm;
            float p0 = __expf(c[0][r] - nm);
            float p1 = __expf(c[1][r] - nm);
            c[0][r] = p0; c[1][r] = p1;
            float sr = p0 + p1;
#pragma unroll
            for (int msk = 1; msk < 16; msk <<= 1)
                sr += __shfl_xor(sr, msk, 32);
            rowL[r] = rowL[r] * al[r] + sr;
        }
#pragma unroll
        for (int n = 0; n < 8; ++n)
#pragma unroll
            for (int r = 0; r < 8; ++r) O[n][r] *= al[r];

        // ---- P (C layout) -> per-wave LDS tile (wave-local, no barrier) ----
        _Float16* Pw = Plds + wave * 16 * LDP;
#pragma unroll
        for (int nt = 0; nt < 2; ++nt)
#pragma unroll
            for (int r = 0; r < 8; ++r)
                Pw[(r + hi * 8) * LDP + nt * 16 + ln] = (_Float16)c[nt][r];

        // ---- O += P * V ----
        const unsigned* Pu = (const unsigned*)(Plds + wave * 16 * LDP);
        Frag16 pf;
#pragma unroll
        for (int vg = 0; vg < 8; ++vg) {
            int kd = vg + hi * 8;
            pf.u[vg] = Pu[ln * (LDP / 2) + kd];
        }
#pragma unroll
        for (int n = 0; n < 8; ++n) {
            Frag16 vf;
            int rb = (n * 16 + ln) * LDV;
#pragma unroll
            for (int vg = 0; vg < 8; ++vg)
                vf.u[vg] = cV[rb + vg + hi * 8];
            O[n] = __builtin_amdgcn_wmma_f32_16x16x32_f16(
                false, pf.v, false, vf.v, (short)0, O[n], false, false);
        }
    };

    constexpr int NSTEP = cfg::SEQ / KT;  // 40
#if USE_ASYNC_LDS
    stage_kv(0, 0);
    for (int st = 0; st < NSTEP; ++st) {
        wait_async0();
        __syncthreads();
        if (st + 1 < NSTEP) stage_kv((st + 1) & 1, (st + 1) * KT);
        compute_step(st & 1);
    }
#else
    load_kv(0);
    store_kv(0);
    for (int st = 0; st < NSTEP - 1; ++st) {
        __syncthreads();
        load_kv((st + 1) * KT);
        compute_step(st & 1);
        store_kv((st + 1) & 1);
    }
    __syncthreads();
    compute_step((NSTEP - 1) & 1);
#endif

    // ---- normalize, store f16 [b*S + q][h*128 + d] ----
    const int b = bh / cfg::H, h = bh % cfg::H;
#pragma unroll
    for (int n = 0; n < 8; ++n)
#pragma unroll
        for (int r = 0; r < 8; ++r) {
            int grow = b * cfg::SEQ + qbase + r + hi * 8;
            int col  = h * cfg::DH + n * 16 + ln;
            AO[(size_t)grow * cfg::DM + col] = (_Float16)(O[n][r] / rowL[r]);
        }
}

// ---------------------------------------------------------------------------
extern "C" void kernel_launch(void* const* d_in, const int* in_sizes, int n_in,
                              void* d_out, int out_size, void* d_ws, size_t ws_size,
                              hipStream_t stream)
{
    (void)in_sizes; (void)n_in; (void)out_size; (void)ws_size;
    const float* hid  = (const float*)d_in[0];
    const float* enc  = (const float*)d_in[1];
    const float* cosb = (const float*)d_in[2];
    const float* sinb = (const float*)d_in[3];
    const float* wqkv = (const float*)d_in[4];  // [3072,9216]
    const float* bqkv = (const float*)d_in[5];
    const float* wout = (const float*)d_in[6];  // [3072,3072]
    const float* bout = (const float*)d_in[7];
    float* out = (float*)d_out;

    char* ws = (char*)d_ws;
    size_t off = 0;
    unsigned* Wp1 = (unsigned*)(ws + off);
    off += (size_t)cfg::N3 * (cfg::DM / 2) * 4;                   // 56.6 MB
    unsigned* Wp2 = (unsigned*)(ws + off);
    off += (size_t)cfg::DM * (cfg::DM / 2) * 4;                   // 18.9 MB
    _Float16* Xh = (_Float16*)(ws + off);
    off += (size_t)cfg::ROWS * cfg::DM * 2;                       // 15.7 MB
    _Float16* qkvh = (_Float16*)(ws + off);
    off += (size_t)cfg::ROWS * cfg::N3 * 2;                       // 47.2 MB
    const size_t hsz = (size_t)cfg::B * cfg::H * cfg::SEQ * cfg::DH * 2;
    _Float16* Qf  = (_Float16*)(ws + off); off += hsz;
    _Float16* Kf  = (_Float16*)(ws + off); off += hsz;
    _Float16* Vtf = (_Float16*)(ws + off); off += hsz;
    _Float16* AOh = (_Float16*)(ws + off);
    off += (size_t)cfg::ROWS * cfg::DM * 2;                       // 15.7 MB

    // 0) one-shot packing: weights (transpose + f16 pair-pack), activations.
    pack_w_kernel<<<(cfg::DM / 32) * (cfg::N3 / 64), 256, 0, stream>>>(
        wqkv, Wp1, cfg::DM, cfg::N3);
    pack_w_kernel<<<(cfg::DM / 32) * (cfg::DM / 64), 256, 0, stream>>>(
        wout, Wp2, cfg::DM, cfg::DM);
    pack_x_kernel<<<(cfg::ROWS * cfg::DM / 4) / 256, 256, 0, stream>>>(
        enc, hid, Xh);

    // 1) QKV projection -> qkv f16, fused bias.
    wmma_gemm_f16<<<dim3(cfg::N3 / 128, cfg::ROWS / 128), 256, 0, stream>>>(
        Xh, Wp1, bqkv, qkvh, cfg::DM, cfg::N3, /*outmode=*/0);

    // 2) LayerNorm + partial RoPE + pack (V transposed).
    ln_rope_pack_kernel<<<(cfg::B * cfg::SEQ * cfg::H) / 8, 256, 0, stream>>>(
        qkvh, cosb, sinb, Qf, Kf, Vtf);

    // 3) Flash attention -> AO f16 (128 Q rows per block).
    attn_fa_kernel<<<cfg::B * cfg::H * (cfg::SEQ / 128), 256, 0, stream>>>(
        Qf, Kf, Vtf, AOh);

    // 4) Output projection, fused bias + (image, text) f32 output remap.
    wmma_gemm_f16<<<dim3(cfg::DM / 128, cfg::ROWS / 128), 256, 0, stream>>>(
        AOh, Wp2, bout, out, cfg::DM, cfg::DM, /*outmode=*/1);
}
